// RKdAngle_70781061039018
// MI455X (gfx1250) — matrix operand order repaired and checked by reading, not compiled, not run on hardware
//
#include <hip/hip_runtime.h>
#include <hip/hip_bf16.h>

// ---------------------------------------------------------------------------
// RKD angle loss, fused for gfx1250 (MI455X, wave32, WMMA f16->f32).
//   N = 384 embeddings, D = 256 dims.
//   One workgroup = (anchor i, j-block of 128, k-block of 128).
//   Phase 1: normalized diffs (f32 math) -> f16 in LDS (264-half padded rows).
//   Phase 2: Gram tiles via v_wmma_f32_16x16x32_f16, 2x4 tile block per wave
//            (waves 4x2 over the 8x8 tile grid) to cut LDS operand traffic
//            ~33% vs a 1x8 strip; smooth-L1 fused in registers.
// ---------------------------------------------------------------------------

typedef __attribute__((ext_vector_type(16))) _Float16 v16h;
typedef __attribute__((ext_vector_type(8)))  _Float16 v8h;
typedef __attribute__((ext_vector_type(8)))  float    v8f;
typedef __attribute__((ext_vector_type(4)))  float    v4f;

#define RKD_N      384
#define RKD_D      256
#define RKD_BLK    128
#define RKD_LSTR   264            // padded row stride in halfs (528B = 132 dw)
#define RKD_KST    8              // K steps of 32 over D=256
#define RKD_LDSB   (4 * RKD_BLK * RKD_LSTR * 2)   // 270336 bytes

__global__ void rkd_zero_kernel(float* acc) {
    if (threadIdx.x == 0 && blockIdx.x == 0) acc[0] = 0.0f;
}

__global__ void rkd_finalize_kernel(const float* acc, float* out) {
    if (threadIdx.x == 0 && blockIdx.x == 0)
        out[0] = acc[0] * (1.0f / (384.0f * 384.0f * 384.0f));
}

// A operand (16x32 f16): lane L<16 holds row M=L, K {k..k+7, k+16..k+23};
// lanes >=16 hold K shifted by 8. Two aligned b128 LDS reads, 32B apart.
__device__ __forceinline__ v16h rkd_loadA(const _Float16* p) {
    v8h lo = *(const v8h*)(p);
    v8h hi = *(const v8h*)(p + 16);
    return __builtin_shufflevector(lo, hi, 0,1,2,3,4,5,6,7,8,9,10,11,12,13,14,15);
}

// B operand (32x16 f16): lane L<16 = column L, K {k..k+7, k+16..k+23};
// lanes >=16 hold K shifted by 8. Two aligned b128 LDS reads, 32B apart.
__device__ __forceinline__ v16h rkd_loadB(const _Float16* p) {
    v8h lo = *(const v8h*)(p);
    v8h hi = *(const v8h*)(p + 16);
    return __builtin_shufflevector(lo, hi, 0,1,2,3,4,5,6,7,8,9,10,11,12,13,14,15);
}

__launch_bounds__(256)
__global__ void rkd_angle_main_kernel(const float* __restrict__ S,
                                      const float* __restrict__ T,
                                      float* __restrict__ acc) {
    extern __shared__ _Float16 lds[];   // [4][128][264] : sj, sk, tj, tk
    __shared__ float red[8];

    const int tid  = threadIdx.x;
    const int wave = tid >> 5;
    const int lane = tid & 31;

    const int bid   = blockIdx.x;
    const int i     = bid / 9;
    const int r9    = bid - i * 9;
    const int bj    = r9 / 3;
    const int bk    = r9 - bj * 3;
    const int jbase = bj * RKD_BLK;
    const int kbase = bk * RKD_BLK;

    // ---------------- Phase 1: normalized diffs into LDS (f16) --------------
    // Wave-constant task region: waves {0,1}->sj, {2,3}->sk, {4,5}->tj, {6,7}->tk
    const int c0 = lane * 8;                 // 8 contiguous floats per lane
    {
        const int which  = wave >> 1;        // 0..3
        const float* E   = (which < 2) ? S : T;
        const int gbase  = (which & 1) ? kbase : jbase;
        _Float16* dstM   = lds + which * (RKD_BLK * RKD_LSTR);
        const int lbase  = (wave & 1) * 64;

        const float* ei = E + i * RKD_D + c0;    // anchor row: loop-invariant
        const v4f a0 = *(const v4f*)(ei);
        const v4f a1 = *(const v4f*)(ei + 4);

        for (int r = 0; r < 64; ++r) {
            const int local = lbase + r;
            const float* er = E + (gbase + local) * RKD_D + c0;
            v4f b0 = *(const v4f*)(er);
            v4f b1 = *(const v4f*)(er + 4);

            float dv[8];
            float ss = 0.0f;
#pragma unroll
            for (int e = 0; e < 4; ++e) {
                float d = a0[e] - b0[e]; dv[e]     = d; ss += d * d;
                float g = a1[e] - b1[e]; dv[e + 4] = g; ss += g * g;
            }
#pragma unroll
            for (int off = 16; off > 0; off >>= 1)
                ss += __shfl_xor(ss, off);

            const float inv = 1.0f / fmaxf(sqrtf(ss), 1e-12f);
            v8h hv;
#pragma unroll
            for (int e = 0; e < 8; ++e) hv[e] = (_Float16)(dv[e] * inv);
            *(v8h*)(dstM + local * RKD_LSTR + c0) = hv;
        }
    }
    __syncthreads();

    // ---------------- Phase 2: Gram tiles via WMMA + fused smooth-L1 --------
    const _Float16* sj = lds;
    const _Float16* sk = lds + 1 * RKD_BLK * RKD_LSTR;
    const _Float16* tj = lds + 2 * RKD_BLK * RKD_LSTR;
    const _Float16* tk = lds + 3 * RKD_BLK * RKD_LSTR;

    const int m    = lane & 15;
    const int hi   = lane >> 4;
    const int aSel = hi << 3;   // A: lanes>=16 hold K 8..15 / 24..31
    const int bSel = hi << 3;   // B: lanes>=16 hold K 8..15 / 24..31

    // Waves tiled 4(rows) x 2(cols) over the 8x8 tile grid; each wave owns
    // a 2x4 block of 16x16 tiles (32x64 of the 128x128 Gram).
    const int trB = (wave & 3) * 2;      // tile rows trB, trB+1
    const int tcB = (wave >> 2) * 4;     // tile cols tcB..tcB+3

    v8f cs[2][4];
    v8f ct[2][4];
#pragma unroll
    for (int r = 0; r < 2; ++r)
#pragma unroll
        for (int c = 0; c < 4; ++c) { cs[r][c] = (v8f){}; ct[r][c] = (v8f){}; }

#pragma unroll
    for (int kt = 0; kt < RKD_KST; ++kt) {
        const int ka = kt * 32 + aSel;
        const int kb = kt * 32 + bSel;

        v16h aS[2], aT[2];
#pragma unroll
        for (int r = 0; r < 2; ++r) {
            const int rowoff = ((trB + r) * 16 + m) * RKD_LSTR;
            aS[r] = rkd_loadA(sj + rowoff + ka);
            aT[r] = rkd_loadA(tj + rowoff + ka);
        }

        v16h bS[4], bT[4];
#pragma unroll
        for (int c = 0; c < 4; ++c) {
            const int coloff = ((tcB + c) * 16 + m) * RKD_LSTR;
            bS[c] = rkd_loadB(sk + coloff + kb);
            bT[c] = rkd_loadB(tk + coloff + kb);
        }

#pragma unroll
        for (int r = 0; r < 2; ++r)
#pragma unroll
            for (int c = 0; c < 4; ++c) {
                cs[r][c] = __builtin_amdgcn_wmma_f32_16x16x32_f16(
                    false, aS[r], false, bS[c], (short)0, cs[r][c], false, false);
                ct[r][c] = __builtin_amdgcn_wmma_f32_16x16x32_f16(
                    false, aT[r], false, bT[c], (short)0, ct[r][c], false, false);
            }
    }

    // Smooth-L1 over the block (layout identical for cs/ct -> elementwise ok)
    float lossAcc = 0.0f;
#pragma unroll
    for (int r = 0; r < 2; ++r)
#pragma unroll
        for (int c = 0; c < 4; ++c)
#pragma unroll
            for (int e = 0; e < 8; ++e) {
                float x = fabsf(cs[r][c][e] - ct[r][c][e]);
                lossAcc += (x < 1.0f) ? (0.5f * x * x) : (x - 0.5f);
            }

#pragma unroll
    for (int off = 16; off > 0; off >>= 1)
        lossAcc += __shfl_xor(lossAcc, off);
    if (lane == 0) red[wave] = lossAcc;
    __syncthreads();
    if (tid == 0) {
        float s = 0.0f;
#pragma unroll
        for (int w = 0; w < 8; ++w) s += red[w];
        atomicAdd(acc, s);
    }
}

extern "C" void kernel_launch(void* const* d_in, const int* in_sizes, int n_in,
                              void* d_out, int out_size, void* d_ws, size_t ws_size,
                              hipStream_t stream) {
    const float* S = (const float*)d_in[0];
    const float* T = (const float*)d_in[1];
    float* acc = (float*)d_ws;
    float* out = (float*)d_out;

    // 264 KB dynamic LDS (CDNA5 WGP allows 320 KB) — raise the runtime cap.
    (void)hipFuncSetAttribute((const void*)rkd_angle_main_kernel,
                              hipFuncAttributeMaxDynamicSharedMemorySize,
                              RKD_LDSB);

    rkd_zero_kernel<<<1, 32, 0, stream>>>(acc);
    rkd_angle_main_kernel<<<dim3(RKD_N * 9), dim3(256), RKD_LDSB, stream>>>(S, T, acc);
    rkd_finalize_kernel<<<1, 32, 0, stream>>>(acc, out);
}
